// ConcatFusionAttention_60352880443933
// MI455X (gfx1250) — compile-verified, matrix operand order checked by reading, not run on hardware
//
#include <hip/hip_runtime.h>

// ---------------------------------------------------------------------------
// ConcatFusionAttention for MI455X (gfx1250, wave32, WMMA f16->f32)
// B=4 T=1024 D=2048 H=16 HD=128 W=256 M=4096 K=8
// Round 3: fix async-builtin pointer types (v4i AS1 src, v4i AS3 dst).
// ---------------------------------------------------------------------------

typedef _Float16 v16h __attribute__((ext_vector_type(16)));
typedef _Float16 v8h  __attribute__((ext_vector_type(8)));
typedef float    v8f  __attribute__((ext_vector_type(8)));
typedef int      a_v4i __attribute__((vector_size(16)));  // matches builtin param

#define CB 4
#define CT 1024
#define CD 2048
#define CH 16
#define CHD 128
#define CW 256
#define CM 4096
#define CK 8

// ---- CDNA5 async global->LDS support (guarded; falls back to sync path) ---
#if defined(__has_builtin)
#if __has_builtin(__builtin_amdgcn_global_load_async_to_lds_b128) && \
    __has_builtin(__builtin_amdgcn_s_wait_asynccnt)
#define HAVE_ASYNC_LDS 1
#endif
#endif
#ifndef HAVE_ASYNC_LDS
#define HAVE_ASYNC_LDS 0
#endif

__device__ __forceinline__ void async_copy_b128(const _Float16* g, _Float16* l) {
#if HAVE_ASYNC_LDS
  __builtin_amdgcn_global_load_async_to_lds_b128(
      (__attribute__((address_space(1))) a_v4i*)g,
      (__attribute__((address_space(3))) a_v4i*)l, 0, 0);
#else
  *(uint4*)l = *(const uint4*)g;  // sync fallback via VGPRs
#endif
}

// -------------------- helpers --------------------------------------------

// Build a 16x32 f16 A/B fragment for one lane from a row-major row pointer.
// Per CDNA5 ISA 16-bit A layout: lane half h holds K = {8h..8h+7, 16+8h..23+8h}.
__device__ __forceinline__ v16h frag_from_row(const _Float16* rowp, int hh) {
  const v8h lo = *(const v8h*)(rowp + 8 * hh);
  const v8h hi = *(const v8h*)(rowp + 16 + 8 * hh);
  v16h f;
#pragma unroll
  for (int e = 0; e < 8; ++e) { f[e] = lo[e]; f[e + 8] = hi[e]; }
  return f;
}

// -------------------- f32 -> f16 cast ------------------------------------

__global__ void cast_f32_f16_kernel(const float* __restrict__ in,
                                    _Float16* __restrict__ out, size_t n) {
  size_t i = (size_t)blockIdx.x * blockDim.x + threadIdx.x;
  if (i < n) out[i] = (_Float16)in[i];
}

// -------------------- WMMA GEMM: C[M,N] = alpha * A[M,K] @ Bm[N,K]^T -----
// 128x128 tile, BK=32, 8 waves (4 along M x 2 along N), double-buffered LDS
// filled by async global->LDS b128 copies (4 per thread per tile).

#define GBM 128
#define GBN 128
#define GBK 32
#define GLDST 40   // padded LDS row stride (f16 elements)

__global__ __launch_bounds__(256) void gemm_f16_abT_kernel(
    const _Float16* __restrict__ A,   // [M,Kd] row-major f16
    const _Float16* __restrict__ Bm,  // [N,Kd] row-major f16 (used transposed)
    float* __restrict__ Cf,           // optional f32 output [M,N]
    _Float16* __restrict__ Ch,        // optional f16 output [M,N]
    int M, int N, int Kd, float alpha) {
  __shared__ __attribute__((aligned(16))) _Float16 Asl[2][GBM * GLDST];
  __shared__ __attribute__((aligned(16))) _Float16 Bsl[2][GBN * GLDST];

  const int tid  = threadIdx.x;
  const int lane = tid & 31;
  const int wave = tid >> 5;
  const int wm   = wave & 3;   // 4 waves along M: 32 rows each
  const int wn   = wave >> 2;  // 2 waves along N: 64 cols each
  const int ln16 = lane & 15;
  const int lh   = lane >> 4;

  const int m0 = blockIdx.y * GBM;
  const int n0 = blockIdx.x * GBN;

  // cooperative fill coords: 256 threads x 16 f16 = 128x32 tile
  const int lr = tid >> 1;            // row 0..127
  const int lc = (tid & 1) * 16;      // col 0 or 16

  const _Float16* gArow = A  + (size_t)(m0 + lr) * Kd + lc;
  const _Float16* gBrow = Bm + (size_t)(n0 + lr) * Kd + lc;
  const int lo = lr * GLDST + lc;     // LDS element offset for this thread

  v8f acc[2][4];
#pragma unroll
  for (int i = 0; i < 2; ++i)
#pragma unroll
    for (int j = 0; j < 4; ++j)
#pragma unroll
      for (int r = 0; r < 8; ++r) acc[i][j][r] = 0.0f;

  auto issue_tile = [&](int kb, int buf) {
    async_copy_b128(gArow + kb,     &Asl[buf][lo]);
    async_copy_b128(gArow + kb + 8, &Asl[buf][lo + 8]);
    async_copy_b128(gBrow + kb,     &Bsl[buf][lo]);
    async_copy_b128(gBrow + kb + 8, &Bsl[buf][lo + 8]);
  };

  const int ntiles = Kd / GBK;
  issue_tile(0, 0);
  if (ntiles > 1) issue_tile(GBK, 1);

  for (int t = 0; t < ntiles; ++t) {
    const int buf = t & 1;
#if HAVE_ASYNC_LDS
    // in-order async completion: with the next tile still in flight (4 ops),
    // asynccnt<=4 means tile t has fully landed in LDS.
    if (t + 1 < ntiles) __builtin_amdgcn_s_wait_asynccnt(4);
    else                __builtin_amdgcn_s_wait_asynccnt(0);
#endif
    __syncthreads();  // all waves' fills visible

    v16h af[2], bf[4];
#pragma unroll
    for (int i = 0; i < 2; ++i)
      af[i] = frag_from_row(&Asl[buf][(32 * wm + 16 * i + ln16) * GLDST], lh);
#pragma unroll
    for (int j = 0; j < 4; ++j)
      bf[j] = frag_from_row(&Bsl[buf][(64 * wn + 16 * j + ln16) * GLDST], lh);

#pragma unroll
    for (int i = 0; i < 2; ++i)
#pragma unroll
      for (int j = 0; j < 4; ++j)
        acc[i][j] = __builtin_amdgcn_wmma_f32_16x16x32_f16(
            false, af[i], false, bf[j], (short)0, acc[i][j], false, false);

    __syncthreads();  // all waves done reading buf before it is refilled
    if (t + 2 < ntiles) issue_tile((t + 2) * GBK, buf);
  }

  // epilogue: C/D layout -> lane (ln16, lh): element (m = r + 8*lh, n = ln16)
#pragma unroll
  for (int i = 0; i < 2; ++i)
#pragma unroll
    for (int j = 0; j < 4; ++j) {
      const int mb = m0 + 32 * wm + 16 * i + 8 * lh;
      const int nn = n0 + 64 * wn + 16 * j + ln16;
#pragma unroll
      for (int r = 0; r < 8; ++r) {
        float vv = acc[i][j][r] * alpha;
        size_t idx = (size_t)(mb + r) * N + nn;
        if (Cf) Cf[idx] = vv;
        if (Ch) Ch[idx] = (_Float16)vv;
      }
    }
}

// -------------------- sliding-window flash attention (WMMA) --------------
// grid = (T/16, H, B), block = 32 (one wave per 16-query tile)

__global__ __launch_bounds__(32) void attn_window_kernel(
    const _Float16* __restrict__ Q,   // [B*T, D] f16 (head h at col h*HD)
    const _Float16* __restrict__ Kt,  // [B*T, D]
    const _Float16* __restrict__ Vt,  // [B*T, D]
    _Float16* __restrict__ ctx_out) { // [B*T, D] f16
  __shared__ __attribute__((aligned(16))) _Float16 Klds[32 * CHD];
  __shared__ __attribute__((aligned(16))) _Float16 Vlds[32 * CHD];
  __shared__ __attribute__((aligned(16))) _Float16 Plds[16 * 32];

  const int lane = threadIdx.x;
  const int ln16 = lane & 15;
  const int lh   = lane >> 4;
  const int q0   = blockIdx.x * 16;
  const int h    = blockIdx.y;
  const int b    = blockIdx.z;

  const size_t headoff = (size_t)b * CT * CD + (size_t)h * CHD;

  // load Q fragments (16 rows x 128 hd = 4 x 16x32 A-frags) from global
  v16h qf[4];
#pragma unroll
  for (int kc = 0; kc < 4; ++kc)
    qf[kc] = frag_from_row(Q + headoff + (size_t)(q0 + ln16) * CD + kc * 32, lh);

  float fm[8], fl[8];
  v8f cacc[8];
#pragma unroll
  for (int r = 0; r < 8; ++r) { fm[r] = -3.0e38f; fl[r] = 0.0f; }
#pragma unroll
  for (int t = 0; t < 8; ++t)
#pragma unroll
    for (int r = 0; r < 8; ++r) cacc[t][r] = 0.0f;

  int jlo = q0 + 15 - (CW - 1);
  if (jlo < 0) jlo = 0;
  jlo &= ~31;
  const int jhi = q0 + 15;
  const float scale = 0.08838834764831845f;  // 1/sqrt(128)

  for (int j0 = jlo; j0 <= jhi; j0 += 32) {
    // stage 32 K rows + 32 V rows into LDS (one row per lane, async b128)
    const _Float16* krow = Kt + headoff + (size_t)(j0 + lane) * CD;
    const _Float16* vrow = Vt + headoff + (size_t)(j0 + lane) * CD;
#pragma unroll
    for (int c = 0; c < 16; ++c) {
      async_copy_b128(krow + c * 8, &Klds[lane * CHD + c * 8]);
      async_copy_b128(vrow + c * 8, &Vlds[lane * CHD + c * 8]);
    }
#if HAVE_ASYNC_LDS
    __builtin_amdgcn_s_wait_asynccnt(0);
#endif
    __syncthreads();  // single wave: orders LDS producer/consumer

    // scores: S[16q x 32j] = Q @ K^T, two 16x16 tiles, K-dim = 128
    v8f s0, s1;
#pragma unroll
    for (int r = 0; r < 8; ++r) { s0[r] = 0.0f; s1[r] = 0.0f; }
#pragma unroll
    for (int kc = 0; kc < 4; ++kc) {
      v16h kb0 = frag_from_row(&Klds[ln16 * CHD + kc * 32], lh);
      v16h kb1 = frag_from_row(&Klds[(16 + ln16) * CHD + kc * 32], lh);
      s0 = __builtin_amdgcn_wmma_f32_16x16x32_f16(false, qf[kc], false, kb0,
                                                  (short)0, s0, false, false);
      s1 = __builtin_amdgcn_wmma_f32_16x16x32_f16(false, qf[kc], false, kb1,
                                                  (short)0, s1, false, false);
    }

    // mask + online softmax update (row stats reduced across 16-lane halves)
    float al[8];
#pragma unroll
    for (int r = 0; r < 8; ++r) {
      const int m  = q0 + r + 8 * lh;
      const int ja = j0 + ln16;
      const int jb = ja + 16;
      float sa = (ja <= m && ja > m - CW) ? s0[r] * scale : -3.0e38f;
      float sb = (jb <= m && jb > m - CW) ? s1[r] * scale : -3.0e38f;
      float mx = fmaxf(sa, sb);
      mx = fmaxf(mx, __shfl_xor(mx, 1, 16));
      mx = fmaxf(mx, __shfl_xor(mx, 2, 16));
      mx = fmaxf(mx, __shfl_xor(mx, 4, 16));
      mx = fmaxf(mx, __shfl_xor(mx, 8, 16));
      float mnew = fmaxf(fm[r], mx);
      float a  = __expf(fm[r] - mnew);
      float p0 = __expf(sa - mnew);
      float p1 = __expf(sb - mnew);
      float rs = p0 + p1;
      rs += __shfl_xor(rs, 1, 16);
      rs += __shfl_xor(rs, 2, 16);
      rs += __shfl_xor(rs, 4, 16);
      rs += __shfl_xor(rs, 8, 16);
      fl[r] = fl[r] * a + rs;
      fm[r] = mnew;
      al[r] = a;
      Plds[(r + 8 * lh) * 32 + ln16]      = (_Float16)p0;
      Plds[(r + 8 * lh) * 32 + 16 + ln16] = (_Float16)p1;
    }
    __syncthreads();

    // ctx(16x128) = ctx*alpha + P(16x32) @ V(32x128)
    v16h pf = frag_from_row(&Plds[ln16 * 32], lh);
#pragma unroll
    for (int t = 0; t < 8; ++t) {
      v8f c = cacc[t];
#pragma unroll
      for (int r = 0; r < 8; ++r) c[r] *= al[r];
      v16h vf;
#pragma unroll
      for (int e = 0; e < 16; ++e) {
        int kk = (e < 8) ? (8 * lh + e) : (16 + 8 * lh + (e - 8));
        vf[e] = Vlds[kk * CHD + t * 16 + ln16];
      }
      cacc[t] = __builtin_amdgcn_wmma_f32_16x16x32_f16(false, pf, false, vf,
                                                       (short)0, c, false, false);
    }
    __syncthreads();
  }

  // epilogue: normalize and store f16 context
#pragma unroll
  for (int t = 0; t < 8; ++t)
#pragma unroll
    for (int r = 0; r < 8; ++r) {
      const int m = q0 + r + 8 * lh;
      float o = cacc[t][r] / fl[r];
      ctx_out[(size_t)(b * CT + m) * CD + h * CHD + t * 16 + ln16] = (_Float16)o;
    }
}

// -------------------- top-k retrieval + weighted gather ------------------
// one wave per (b,t) row; block = 256 (8 waves), grid = R/8

__global__ __launch_bounds__(256) void topk_mem_kernel(
    const float* __restrict__ sim,    // [R, M] (already scaled by 1/sqrt(D))
    const float* __restrict__ vmem,   // [M, D] f32
    float* __restrict__ omem_f32,     // [R, D]
    _Float16* __restrict__ omem_f16) {// [R, D]
  __shared__ float svals[8][256];
  __shared__ int   sidx[8][256];
  __shared__ float wsel[8][CK];
  __shared__ int   isel[8][CK];

  const int lane = threadIdx.x & 31;
  const int wave = threadIdx.x >> 5;
  const int row  = blockIdx.x * 8 + wave;
  const float* srow = sim + (size_t)row * CM;

  float tv[CK];
  int   ti[CK];
#pragma unroll
  for (int s = 0; s < CK; ++s) { tv[s] = -3.0e38f; ti[s] = 0; }

  for (int j = lane; j < CM; j += 32) {
    float v = srow[j];
    if (v > tv[CK - 1]) {
      tv[CK - 1] = v; ti[CK - 1] = j;
#pragma unroll
      for (int s = CK - 1; s > 0; --s)
        if (tv[s] > tv[s - 1]) {
          float fv = tv[s]; tv[s] = tv[s - 1]; tv[s - 1] = fv;
          int   iv = ti[s]; ti[s] = ti[s - 1]; ti[s - 1] = iv;
        }
    }
  }
#pragma unroll
  for (int s = 0; s < CK; ++s) {
    svals[wave][lane * CK + s] = tv[s];
    sidx[wave][lane * CK + s]  = ti[s];
  }
  __syncthreads();

  if (lane == 0) {
    float bv[CK]; int bi[CK];
#pragma unroll
    for (int s = 0; s < CK; ++s) { bv[s] = -3.0e38f; bi[s] = 0; }
    for (int c = 0; c < 256; ++c) {
      float v = svals[wave][c];
      if (v > bv[CK - 1]) {
        bv[CK - 1] = v; bi[CK - 1] = sidx[wave][c];
#pragma unroll
        for (int s = CK - 1; s > 0; --s)
          if (bv[s] > bv[s - 1]) {
            float fv = bv[s]; bv[s] = bv[s - 1]; bv[s - 1] = fv;
            int   iv = bi[s]; bi[s] = bi[s - 1]; bi[s - 1] = iv;
          }
      }
    }
    float mx = bv[0], sum = 0.0f, e[CK];
#pragma unroll
    for (int s = 0; s < CK; ++s) { e[s] = __expf(bv[s] - mx); sum += e[s]; }
    float inv = 1.0f / sum;
#pragma unroll
    for (int s = 0; s < CK; ++s) { wsel[wave][s] = e[s] * inv; isel[wave][s] = bi[s]; }
  }
  __syncthreads();

  float wk[CK]; size_t vo[CK];
#pragma unroll
  for (int s = 0; s < CK; ++s) {
    wk[s] = wsel[wave][s];
    vo[s] = (size_t)isel[wave][s] * CD;
  }
  for (int d = lane; d < CD; d += 32) {
    float a = 0.0f;
#pragma unroll
    for (int s = 0; s < CK; ++s) a += wk[s] * vmem[vo[s] + d];
    omem_f32[(size_t)row * CD + d] = a;
    omem_f16[(size_t)row * CD + d] = (_Float16)a;
  }
}

// -------------------- gated fusion ---------------------------------------
// out = o_local + sigmoid(o_local . gate_w + gate_b) * o_mem ; wave per row

__global__ __launch_bounds__(256) void fuse_kernel(
    const float* __restrict__ o_local, const float* __restrict__ o_mem,
    const float* __restrict__ gate_w, const float* __restrict__ gate_b,
    float* __restrict__ out) {
  const int lane = threadIdx.x & 31;
  const int wave = threadIdx.x >> 5;
  const int row  = blockIdx.x * 8 + wave;
  const float* ol = o_local + (size_t)row * CD;
  const float* om = o_mem   + (size_t)row * CD;

  float p = 0.0f;
  for (int d = lane; d < CD; d += 32) p += ol[d] * gate_w[d];
  p += __shfl_xor(p, 16, 32);
  p += __shfl_xor(p, 8, 32);
  p += __shfl_xor(p, 4, 32);
  p += __shfl_xor(p, 2, 32);
  p += __shfl_xor(p, 1, 32);
  const float g = 1.0f / (1.0f + __expf(-(p + gate_b[0])));

  for (int d = lane; d < CD; d += 32)
    out[(size_t)row * CD + d] = ol[d] + g * om[d];
}

// -------------------- host launch ----------------------------------------

extern "C" void kernel_launch(void* const* d_in, const int* in_sizes, int n_in,
                              void* d_out, int out_size, void* d_ws, size_t ws_size,
                              hipStream_t stream) {
  (void)in_sizes; (void)n_in; (void)out_size; (void)ws_size;
  const float* hs = (const float*)d_in[0];
  const float* wq = (const float*)d_in[1];
  const float* wk = (const float*)d_in[2];
  const float* wv = (const float*)d_in[3];
  const float* wo = (const float*)d_in[4];
  const float* gw = (const float*)d_in[5];
  const float* gb = (const float*)d_in[6];
  const float* mb = (const float*)d_in[7];
  float* out = (float*)d_out;

  const int R = CB * CT;  // 4096 token rows

  char* wsb = (char*)d_ws;
  size_t off = 0;
  auto take = [&](size_t bytes) -> void* {
    void* p = wsb + off;
    off = (off + bytes + 255) & ~(size_t)255;
    return p;
  };
  _Float16* hs16    = (_Float16*)take((size_t)R  * CD * 2);
  _Float16* wq16    = (_Float16*)take((size_t)CD * CD * 2);
  _Float16* wk16    = (_Float16*)take((size_t)CD * CD * 2);
  _Float16* wv16    = (_Float16*)take((size_t)CD * CD * 2);
  _Float16* wo16    = (_Float16*)take((size_t)CD * CD * 2);
  _Float16* mb16    = (_Float16*)take((size_t)CM * CD * 2);
  _Float16* q16     = (_Float16*)take((size_t)R  * CD * 2);
  _Float16* k16     = (_Float16*)take((size_t)R  * CD * 2);
  _Float16* vv16    = (_Float16*)take((size_t)R  * CD * 2);
  _Float16* kmem16  = (_Float16*)take((size_t)CM * CD * 2);
  float*    vmem32  = (float*)   take((size_t)CM * CD * 4);
  float*    sim32   = (float*)   take((size_t)R  * CM * 4);
  float*    omemp32 = (float*)   take((size_t)R  * CD * 4);
  _Float16* omemp16 = (_Float16*)take((size_t)R  * CD * 2);
  _Float16* ctx16   = (_Float16*)take((size_t)R  * CD * 2);
  float*    oloc32  = (float*)   take((size_t)R  * CD * 4);
  float*    omem32  = (float*)   take((size_t)R  * CD * 4);

  auto cast = [&](const float* src, _Float16* dst, size_t n) {
    cast_f32_f16_kernel<<<(unsigned)((n + 255) / 256), 256, 0, stream>>>(src, dst, n);
  };
  cast(hs, hs16, (size_t)R * CD);
  cast(wq, wq16, (size_t)CD * CD);
  cast(wk, wk16, (size_t)CD * CD);
  cast(wv, wv16, (size_t)CD * CD);
  cast(wo, wo16, (size_t)CD * CD);
  cast(mb, mb16, (size_t)CM * CD);

  auto gemm = [&](const _Float16* A, const _Float16* Bm, float* Cf, _Float16* Ch,
                  int M, int N, int Kd, float alpha) {
    dim3 grid(N / GBN, M / GBM);
    gemm_f16_abT_kernel<<<grid, 256, 0, stream>>>(A, Bm, Cf, Ch, M, N, Kd, alpha);
  };

  // projections
  gemm(hs16, wq16, nullptr, q16,  R,  CD, CD, 1.0f);
  gemm(hs16, wk16, nullptr, k16,  R,  CD, CD, 1.0f);
  gemm(hs16, wv16, nullptr, vv16, R,  CD, CD, 1.0f);
  gemm(mb16, wk16, nullptr, kmem16, CM, CD, CD, 1.0f);
  gemm(mb16, wv16, vmem32, nullptr, CM, CD, CD, 1.0f);
  // sim = hs @ k_mem^T / sqrt(D)
  gemm(hs16, kmem16, sim32, nullptr, R, CM, CD, 0.022097086912079608f);

  // local sliding-window attention -> ctx (f16)
  attn_window_kernel<<<dim3(CT / 16, CH, CB), 32, 0, stream>>>(q16, k16, vv16, ctx16);

  // memory path: top-8 softmax gather -> o_mem_pre
  topk_mem_kernel<<<R / 8, 256, 0, stream>>>(sim32, vmem32, omemp32, omemp16);

  // output projections
  gemm(ctx16,   wo16, oloc32, nullptr, R, CD, CD, 1.0f);
  gemm(omemp16, wo16, omem32, nullptr, R, CD, CD, 1.0f);

  // gated fusion
  fuse_kernel<<<R / 8, 256, 0, stream>>>(oloc32, omem32, gw, gb, out);
}
